// MoE_38285338477197
// MI455X (gfx1250) — matrix operand order repaired and checked by reading, not compiled, not run on hardware
//
#include <hip/hip_runtime.h>
#include <hip/hip_bf16.h>
#include <stdint.h>

#define N_TOK 4096
#define DDIM  1024
#define HDIM  4096
#define NEXP  8
#define MAXSLOTS (N_TOK + NEXP * 16)   // 4224: each expert group padded to 16

typedef __attribute__((ext_vector_type(16))) __bf16   v16bf;
typedef __attribute__((ext_vector_type(8)))  float    v8f;
typedef __attribute__((ext_vector_type(4)))  uint32_t v4u;

__device__ __forceinline__ uint32_t f2bf1(float f) {
    union { float f; uint32_t u; } c; c.f = f;
    uint32_t u = c.u;
    return (u + 0x7FFFu + ((u >> 16) & 1u)) >> 16;   // RNE
}
__device__ __forceinline__ uint32_t f2bf2(float lo, float hi) {
    return f2bf1(lo) | (f2bf1(hi) << 16);
}

// ---------------- init: clear counters, poison perm with -1 ----------------
__global__ void init_kernel(int* __restrict__ perm, int* __restrict__ counts,
                            int* __restrict__ cursor) {
    int i = blockIdx.x * blockDim.x + threadIdx.x;
    if (i < MAXSLOTS) perm[i] = -1;
    if (i < NEXP) { counts[i] = 0; cursor[i] = 0; }
}

// ---------------- gate: one wave per token, 8 logits, argmax ----------------
__global__ __launch_bounds__(256)
void gate_kernel(const float* __restrict__ x, const float* __restrict__ gw,
                 const float* __restrict__ gb, int* __restrict__ top1,
                 int* __restrict__ counts) {
    int wave = threadIdx.x >> 5, lane = threadIdx.x & 31;
    int t = blockIdx.x * 8 + wave;
    if (t >= N_TOK) return;

    float acc[NEXP];
#pragma unroll
    for (int e = 0; e < NEXP; ++e) acc[e] = 0.f;

    for (int d = lane; d < DDIM; d += 32) {
        float xv = x[(size_t)t * DDIM + d];
        const float4* g4 = (const float4*)(gw + (size_t)d * NEXP);
        float4 a = g4[0], b = g4[1];
        acc[0] += xv * a.x; acc[1] += xv * a.y;
        acc[2] += xv * a.z; acc[3] += xv * a.w;
        acc[4] += xv * b.x; acc[5] += xv * b.y;
        acc[6] += xv * b.z; acc[7] += xv * b.w;
    }
#pragma unroll
    for (int e = 0; e < NEXP; ++e)
        for (int off = 16; off >= 1; off >>= 1)
            acc[e] += __shfl_xor(acc[e], off, 32);

    if (lane == 0) {
        int best = 0; float bv = acc[0] + gb[0];
        for (int e = 1; e < NEXP; ++e) {
            float v = acc[e] + gb[e];
            if (v > bv) { bv = v; best = e; }   // first max, like jnp.argmax
        }
        top1[t] = best;
        atomicAdd(&counts[best], 1);
    }
}

// ---------------- offsets: 16-aligned prefix sum over 8 experts ----------------
__global__ void offsets_kernel(const int* __restrict__ counts,
                               int* __restrict__ offs, int* __restrict__ meta) {
    if (threadIdx.x == 0 && blockIdx.x == 0) {
        int acc = 0;
        for (int e = 0; e < NEXP; ++e) {
            offs[e] = acc;
            acc += (counts[e] + 15) & ~15;
        }
        offs[NEXP] = acc;
        meta[0] = acc >> 4;   // number of 16-row M tiles
    }
}

// ---------------- scatter tokens into expert buckets ----------------
__global__ void scatter_kernel(const int* __restrict__ top1,
                               const int* __restrict__ offs,
                               int* __restrict__ cursor, int* __restrict__ perm) {
    int t = blockIdx.x * blockDim.x + threadIdx.x;
    if (t < N_TOK) {
        int e = top1[t];
        int p = atomicAdd(&cursor[e], 1);
        perm[offs[e] + p] = t;
    }
}

// LDS tiles (double buffered). A: 16 rows x 32 k, B transposed: 64 n x 32 k,
// both with 80-byte row stride so fragments are two aligned ds_load_b128 each
// and 16 lanes spread over all 64 banks.
#define ASTR 40
#define BSTR 40

// ---------------- GEMM1: H = relu(X @ W1[e] + b1[e]), bf16 out ----------------
__global__ __launch_bounds__(128)
void ffn1_kernel(const float* __restrict__ x, const float* __restrict__ W1,
                 const float* __restrict__ b1, const int* __restrict__ perm,
                 const int* __restrict__ offs, const int* __restrict__ meta,
                 unsigned short* __restrict__ Hbuf) {
    int tile = blockIdx.x;
    if (tile >= meta[0]) return;
    int slot0 = tile * 16;
    int e = 0;
    while (e < NEXP - 1 && slot0 >= offs[e + 1]) ++e;
    int n0 = blockIdx.y * 64;

    __shared__ __align__(16) unsigned short sA[2][16][ASTR];
    __shared__ __align__(16) unsigned short sBt[2][64][BSTR];
    __shared__ int sTok[16];

    int tid = threadIdx.x;
    int wave = tid >> 5, lane = tid & 31;
    if (tid < 16) sTok[tid] = perm[slot0 + tid];
    __syncthreads();

    const float* Wbase = W1 + (size_t)e * DDIM * HDIM;

    // hoisted staging coordinates
    int ar = tid >> 3, ac = (tid & 7) * 4;
    int tokA = sTok[ar];
    const float* pA = x + (size_t)(tokA < 0 ? 0 : tokA) * DDIM + ac;
    float amsk = (tokA < 0) ? 0.f : 1.f;

    int np[4], kk[4];
    const float* pB[4];
#pragma unroll
    for (int j = 0; j < 4; ++j) {
        int u = tid + 128 * j;
        np[j] = (u & 31) * 2;
        kk[j] = (u >> 5) * 2;
        pB[j] = Wbase + (size_t)kk[j] * HDIM + n0 + np[j];
    }
    int m = lane & 15, kh = lane >> 4;
    int cn = wave * 16 + m;

    v8f acc = {};

    // prologue: stage chunk 0 into buffer 0
    {
        float4 av = *(const float4*)pA;
        float2 rB[8];
#pragma unroll
        for (int j = 0; j < 4; ++j) {
            rB[2 * j]     = *(const float2*)pB[j];
            rB[2 * j + 1] = *(const float2*)(pB[j] + HDIM);
        }
        uint2 pa; pa.x = f2bf2(av.x * amsk, av.y * amsk);
        pa.y = f2bf2(av.z * amsk, av.w * amsk);
        *(uint2*)&sA[0][ar][ac] = pa;
#pragma unroll
        for (int j = 0; j < 4; ++j) {
            *(uint32_t*)&sBt[0][np[j]][kk[j]]     = f2bf2(rB[2 * j].x, rB[2 * j + 1].x);
            *(uint32_t*)&sBt[0][np[j] + 1][kk[j]] = f2bf2(rB[2 * j].y, rB[2 * j + 1].y);
        }
    }
    __syncthreads();

    const int NC = DDIM / 32;
    int p = 0;
    for (int c = 0; c < NC; ++c) {
        bool more = (c + 1 < NC);
        int k0n = (c + 1) * 32;
        float4 avn = make_float4(0.f, 0.f, 0.f, 0.f);
        float2 rBn[8];
        if (more) {   // issue next chunk's global loads before the WMMA
            avn = *(const float4*)(pA + k0n);
#pragma unroll
            for (int j = 0; j < 4; ++j) {
                const float* q = pB[j] + (size_t)k0n * HDIM;
                rBn[2 * j]     = *(const float2*)q;
                rBn[2 * j + 1] = *(const float2*)(q + HDIM);
            }
        }
        // compute on buffer p
        union { v16bf v; v4u q[2]; } a, b;
        a.q[0] = *(const v4u*)&sA[p][m][kh * 8];
        a.q[1] = *(const v4u*)&sA[p][m][16 + kh * 8];
        b.q[0] = *(const v4u*)&sBt[p][cn][kh * 8];
        b.q[1] = *(const v4u*)&sBt[p][cn][16 + kh * 8];
        acc = __builtin_amdgcn_wmma_f32_16x16x32_bf16(
            false, a.v, false, b.v, (short)0, acc, false, false);
        if (more) {   // convert + store next chunk into other buffer
            uint2 pa; pa.x = f2bf2(avn.x * amsk, avn.y * amsk);
            pa.y = f2bf2(avn.z * amsk, avn.w * amsk);
            *(uint2*)&sA[p ^ 1][ar][ac] = pa;
#pragma unroll
            for (int j = 0; j < 4; ++j) {
                *(uint32_t*)&sBt[p ^ 1][np[j]][kk[j]]     = f2bf2(rBn[2 * j].x, rBn[2 * j + 1].x);
                *(uint32_t*)&sBt[p ^ 1][np[j] + 1][kk[j]] = f2bf2(rBn[2 * j].y, rBn[2 * j + 1].y);
            }
        }
        __syncthreads();
        p ^= 1;
    }

    // epilogue: bias + relu -> bf16 H buffer. D layout: VGPR r -> M = r + 8*(lane>=16)
    int nn = n0 + wave * 16 + (lane & 15);
    int mbase = (lane >> 4) * 8;
    float bias = b1[(size_t)e * HDIM + nn];
#pragma unroll
    for (int r = 0; r < 8; ++r) {
        float v = acc[r] + bias;
        v = v > 0.f ? v : 0.f;
        Hbuf[(size_t)(slot0 + mbase + r) * HDIM + nn] = (unsigned short)f2bf1(v);
    }
}

// ---------------- GEMM2: out[tok] = H @ W2[e] + b2[e] ----------------
__global__ __launch_bounds__(128)
void ffn2_kernel(const unsigned short* __restrict__ Hbuf,
                 const float* __restrict__ W2, const float* __restrict__ b2,
                 const int* __restrict__ perm, const int* __restrict__ offs,
                 const int* __restrict__ meta, float* __restrict__ out) {
    int tile = blockIdx.x;
    if (tile >= meta[0]) return;
    int slot0 = tile * 16;
    int e = 0;
    while (e < NEXP - 1 && slot0 >= offs[e + 1]) ++e;
    int n0 = blockIdx.y * 64;

    __shared__ __align__(16) unsigned short sA[2][16][ASTR];
    __shared__ __align__(16) unsigned short sBt[2][64][BSTR];
    __shared__ int sTok[16];

    int tid = threadIdx.x;
    int wave = tid >> 5, lane = tid & 31;
    if (tid < 16) sTok[tid] = perm[slot0 + tid];
    __syncthreads();

    const float* Wbase = W2 + (size_t)e * HDIM * DDIM;

    int ar = tid >> 3, ac = (tid & 7) * 4;
    const unsigned short* pH = Hbuf + (size_t)(slot0 + ar) * HDIM + ac;

    int np[4], kk[4];
    const float* pB[4];
#pragma unroll
    for (int j = 0; j < 4; ++j) {
        int u = tid + 128 * j;
        np[j] = (u & 31) * 2;
        kk[j] = (u >> 5) * 2;
        pB[j] = Wbase + (size_t)kk[j] * DDIM + n0 + np[j];
    }
    int m = lane & 15, kh = lane >> 4;
    int cn = wave * 16 + m;

    v8f acc = {};

    // prologue: stage chunk 0 into buffer 0.
    // A operand is already bf16 -> use CDNA5 async global->LDS copy (ASYNCcnt).
    {
        uint32_t lds0 = (uint32_t)(uintptr_t)&sA[0][ar][ac];
        asm volatile("global_load_async_to_lds_b64 %0, %1, off"
                     :: "v"(lds0), "v"(pH) : "memory");
        float2 rB[8];
#pragma unroll
        for (int j = 0; j < 4; ++j) {
            rB[2 * j]     = *(const float2*)pB[j];
            rB[2 * j + 1] = *(const float2*)(pB[j] + DDIM);
        }
#pragma unroll
        for (int j = 0; j < 4; ++j) {
            *(uint32_t*)&sBt[0][np[j]][kk[j]]     = f2bf2(rB[2 * j].x, rB[2 * j + 1].x);
            *(uint32_t*)&sBt[0][np[j] + 1][kk[j]] = f2bf2(rB[2 * j].y, rB[2 * j + 1].y);
        }
        asm volatile("s_wait_asynccnt 0" ::: "memory");
    }
    __syncthreads();

    const int NC = HDIM / 32;
    int p = 0;
    for (int c = 0; c < NC; ++c) {
        bool more = (c + 1 < NC);
        int k0n = (c + 1) * 32;
        float2 rBn[8];
        if (more) {
            // async A copy for next chunk straight into the inactive buffer
            uint32_t ldsn = (uint32_t)(uintptr_t)&sA[p ^ 1][ar][ac];
            const unsigned short* gn = pH + k0n;
            asm volatile("global_load_async_to_lds_b64 %0, %1, off"
                         :: "v"(ldsn), "v"(gn) : "memory");
            // B next chunk into registers
#pragma unroll
            for (int j = 0; j < 4; ++j) {
                const float* q = pB[j] + (size_t)k0n * DDIM;
                rBn[2 * j]     = *(const float2*)q;
                rBn[2 * j + 1] = *(const float2*)(q + DDIM);
            }
        }
        union { v16bf v; v4u q[2]; } a, b;
        a.q[0] = *(const v4u*)&sA[p][m][kh * 8];
        a.q[1] = *(const v4u*)&sA[p][m][16 + kh * 8];
        b.q[0] = *(const v4u*)&sBt[p][cn][kh * 8];
        b.q[1] = *(const v4u*)&sBt[p][cn][16 + kh * 8];
        acc = __builtin_amdgcn_wmma_f32_16x16x32_bf16(
            false, a.v, false, b.v, (short)0, acc, false, false);
        if (more) {
#pragma unroll
            for (int j = 0; j < 4; ++j) {
                *(uint32_t*)&sBt[p ^ 1][np[j]][kk[j]]     = f2bf2(rBn[2 * j].x, rBn[2 * j + 1].x);
                *(uint32_t*)&sBt[p ^ 1][np[j] + 1][kk[j]] = f2bf2(rBn[2 * j].y, rBn[2 * j + 1].y);
            }
            asm volatile("s_wait_asynccnt 0" ::: "memory");
        }
        __syncthreads();
        p ^= 1;
    }

    int nn = n0 + wave * 16 + (lane & 15);
    int mbase = (lane >> 4) * 8;
    float bias = b2[(size_t)e * DDIM + nn];
#pragma unroll
    for (int r = 0; r < 8; ++r) {
        int tok = sTok[mbase + r];
        if (tok >= 0) out[(size_t)tok * DDIM + nn] = acc[r] + bias;
    }
}

// ---------------- launcher ----------------
extern "C" void kernel_launch(void* const* d_in, const int* in_sizes, int n_in,
                              void* d_out, int out_size, void* d_ws, size_t ws_size,
                              hipStream_t stream) {
    (void)in_sizes; (void)n_in; (void)out_size; (void)ws_size;
    const float* x  = (const float*)d_in[0];
    const float* gw = (const float*)d_in[1];
    const float* gb = (const float*)d_in[2];
    const float* W1 = (const float*)d_in[3];
    const float* b1 = (const float*)d_in[4];
    const float* W2 = (const float*)d_in[5];
    const float* b2 = (const float*)d_in[6];
    float* out = (float*)d_out;

    char* base   = (char*)d_ws;
    int* top1    = (int*)base;                 // 4096
    int* perm    = top1 + N_TOK;               // 4224
    int* counts  = perm + MAXSLOTS;            // 8
    int* offs    = counts + 8;                 // 9 (padded to 16)
    int* cursor  = offs + 16;                  // 8
    int* meta    = cursor + 8;                 // 4
    unsigned short* Hbuf = (unsigned short*)(base + 65536);  // 4224*4096 bf16 = 34.6 MB

    init_kernel<<<(MAXSLOTS + 255) / 256, 256, 0, stream>>>(perm, counts, cursor);
    gate_kernel<<<N_TOK / 8, 256, 0, stream>>>(x, gw, gb, top1, counts);
    offsets_kernel<<<1, 32, 0, stream>>>(counts, offs, meta);
    scatter_kernel<<<N_TOK / 256, 256, 0, stream>>>(top1, offs, cursor, perm);
    ffn1_kernel<<<dim3(MAXSLOTS / 16, HDIM / 64), 128, 0, stream>>>(
        x, W1, b1, perm, offs, meta, Hbuf);
    ffn2_kernel<<<dim3(MAXSLOTS / 16, DDIM / 64), 128, 0, stream>>>(
        Hbuf, W2, b2, perm, offs, meta, out);
}
